// FAConv_layer_72688026518109
// MI455X (gfx1250) — matrix operand order repaired and checked by reading, not compiled, not run on hardware
//
#include <hip/hip_runtime.h>
#include <hip/hip_bf16.h>

#define D_FEAT 256
#define EPS_FA 0.1f
#define EPS_LN 1e-5f

typedef __attribute__((ext_vector_type(16))) __bf16 v16bf;
typedef __attribute__((ext_vector_type(8)))  float  v8f;

// ---------------------------------------------------------------------------
// 1) out = EPS_FA * node_0   (also resets accumulator deterministically)
// ---------------------------------------------------------------------------
__global__ __launch_bounds__(256) void fa_init_kernel(const float4* __restrict__ n0,
                                                      float4* __restrict__ out, int n4) {
    int i = blockIdx.x * 256 + threadIdx.x;
    if (i < n4) {
        float4 v = n0[i];
        out[i] = make_float4(EPS_FA * v.x, EPS_FA * v.y, EPS_FA * v.z, EPS_FA * v.w);
    }
}

// ---------------------------------------------------------------------------
// 2) a_l = node @ att_l, a_r = node @ att_r via v_wmma_f32_16x16x32_bf16,
//    using the bf16x3 split (x ~= hi + lo) for ~fp32 accuracy:
//      acc += Ah*Bh + Al*Bh + Ah*Bl      (Al*Bl term is ~2^-16, dropped)
//    One wave (32 lanes) per tile of 16 nodes. B has att_l in col 0, att_r
//    in col 1, zeros elsewhere. 8 K-steps of 32 cover D = 256.
// ---------------------------------------------------------------------------
__global__ __launch_bounds__(32) void fa_att_kernel(const float* __restrict__ node,
                                                    const float* __restrict__ att_l,
                                                    const float* __restrict__ att_r,
                                                    float* __restrict__ outl,
                                                    float* __restrict__ outr, int Nn) {
    const int lane = threadIdx.x;          // 0..31
    const int tile = blockIdx.x;           // tile of 16 nodes
    const int row  = lane & 15;
    const int hi   = lane >> 4;            // 0: lanes 0-15, 1: lanes 16-31

    int ridx = tile * 16 + row;
    if (ridx >= Nn) ridx = Nn - 1;         // keep EXEC full for WMMA
    const float* __restrict__ arow = node + (size_t)ridx * D_FEAT;

    v8f acc = {};
    for (int kb = 0; kb < D_FEAT; kb += 32) {
        // ---- A matrix: 16x32 bf16 (ISA 7.12.2 layout)
        // lane holds K = kb+hi*8 .. +7  and  kb+16+hi*8 .. +7
        const int ka = kb + hi * 8;
        float fa[16];
        *(float4*)(fa + 0)  = *(const float4*)(arow + ka);
        *(float4*)(fa + 4)  = *(const float4*)(arow + ka + 4);
        *(float4*)(fa + 8)  = *(const float4*)(arow + ka + 16);
        *(float4*)(fa + 12) = *(const float4*)(arow + ka + 20);
        v16bf a_hi, a_lo;
#pragma unroll
        for (int i = 0; i < 16; ++i) {
            __bf16 h = (__bf16)fa[i];
            a_hi[i] = h;
            a_lo[i] = (__bf16)(fa[i] - (float)h);
        }

        // ---- B matrix: 32x16 bf16; column N = lane&15. col0=att_l, col1=att_r
        // lane holds K = kb+hi*8 .. +7  and  kb+16+hi*8 .. +7 (same as A)
        v16bf b_hi = {}, b_lo = {};
        if (row < 2) {
            const float* __restrict__ s = (row == 0) ? att_l : att_r;
            const int kbB = kb + hi * 16;
#pragma unroll
            for (int i = 0; i < 16; ++i) {
                float x = s[kb + hi * 8 + i + (i >= 8 ? 8 : 0)];
                __bf16 h = (__bf16)x;
                b_hi[i] = h;
                b_lo[i] = (__bf16)(x - (float)h);
            }
        }

        acc = __builtin_amdgcn_wmma_f32_16x16x32_bf16(false, a_hi, false, b_hi,
                                                      (short)0, acc, false, false);
        acc = __builtin_amdgcn_wmma_f32_16x16x32_bf16(false, a_lo, false, b_hi,
                                                      (short)0, acc, false, false);
        acc = __builtin_amdgcn_wmma_f32_16x16x32_bf16(false, a_hi, false, b_lo,
                                                      (short)0, acc, false, false);
    }

    // C layout: VGPR v, lanes 0-15 -> M=v (N=lane), lanes 16-31 -> M=v+8 (N=lane-16)
    const int base = tile * 16 + hi * 8;
    if (row == 0) {
#pragma unroll
        for (int v = 0; v < 8; ++v)
            if (base + v < Nn) outl[base + v] = acc[v];
    } else if (row == 1) {
#pragma unroll
        for (int v = 0; v < 8; ++v)
            if (base + v < Nn) outr[base + v] = acc[v];
    }
}

// ---------------------------------------------------------------------------
// 3) per-edge coefficient: coef = tanh(a_l[src] + a_r[dst]) * edge_attr.
//    Also prefetch the source node row (gathered by the scatter kernel next)
//    into L2 -- node is 51 MB, fully L2-resident on MI455X's 192 MB L2.
// ---------------------------------------------------------------------------
__global__ __launch_bounds__(256) void fa_coef_kernel(const long long* __restrict__ ei,
                                                      const float* __restrict__ eattr,
                                                      const float* __restrict__ al,
                                                      const float* __restrict__ ar,
                                                      const float* __restrict__ node,
                                                      float* __restrict__ coef, int E) {
    int e = blockIdx.x * 256 + threadIdx.x;
    if (e < E) {
        long long s = ei[e];
        long long d = ei[(size_t)E + e];
        coef[e] = tanhf(al[s] + ar[d]) * eattr[e];
        // warm L2 with the 1 KB source row (8 x 128B lines) ahead of scatter
        const char* p = (const char*)(node + (size_t)s * D_FEAT);
#pragma unroll
        for (int i = 0; i < 8; ++i)
            __builtin_prefetch(p + i * 128, 0, 1);
    }
}

// ---------------------------------------------------------------------------
// 4) scatter: out[dst] += node[src] * coef   (64 threads per edge, float4 each)
// ---------------------------------------------------------------------------
__global__ __launch_bounds__(256) void fa_scatter_kernel(const float* __restrict__ node,
                                                         const long long* __restrict__ ei,
                                                         const float* __restrict__ coef,
                                                         float* __restrict__ out, int E) {
    int t = blockIdx.x * 256 + threadIdx.x;
    int e = t >> 6;            // 64 threads per edge
    int l = t & 63;            // feature chunk (4 floats each)
    if (e >= E) return;
    float c = coef[e];
    long long s = ei[e];
    long long d = ei[(size_t)E + e];
    float4 v = *(const float4*)(node + (size_t)s * D_FEAT + l * 4);
    float* o = out + (size_t)d * D_FEAT + l * 4;
    __hip_atomic_fetch_add(o + 0, c * v.x, __ATOMIC_RELAXED, __HIP_MEMORY_SCOPE_AGENT);
    __hip_atomic_fetch_add(o + 1, c * v.y, __ATOMIC_RELAXED, __HIP_MEMORY_SCOPE_AGENT);
    __hip_atomic_fetch_add(o + 2, c * v.z, __ATOMIC_RELAXED, __HIP_MEMORY_SCOPE_AGENT);
    __hip_atomic_fetch_add(o + 3, c * v.w, __ATOMIC_RELAXED, __HIP_MEMORY_SCOPE_AGENT);
}

// ---------------------------------------------------------------------------
// 5) per-row LayerNorm + ReLU, in place on out. One 256-thread block per node.
// ---------------------------------------------------------------------------
__global__ __launch_bounds__(256) void fa_ln_relu_kernel(float* __restrict__ out,
                                                         const float* __restrict__ w,
                                                         const float* __restrict__ b) {
    __shared__ float s_sum[8];
    __shared__ float s_sq[8];
    __shared__ float s_mean, s_inv;

    const int t = threadIdx.x;
    float* rowp = out + (size_t)blockIdx.x * D_FEAT;
    float x = rowp[t];

    float s = x, q = x * x;
#pragma unroll
    for (int m = 16; m >= 1; m >>= 1) {
        s += __shfl_xor(s, m, 32);
        q += __shfl_xor(q, m, 32);
    }
    const int wid = t >> 5;
    if ((t & 31) == 0) { s_sum[wid] = s; s_sq[wid] = q; }
    __syncthreads();
    if (t == 0) {
        float S = 0.f, Q = 0.f;
#pragma unroll
        for (int i = 0; i < 8; ++i) { S += s_sum[i]; Q += s_sq[i]; }
        float mean = S * (1.0f / D_FEAT);
        float var  = Q * (1.0f / D_FEAT) - mean * mean;
        s_mean = mean;
        s_inv  = rsqrtf(var + EPS_LN);
    }
    __syncthreads();
    float y = (x - s_mean) * s_inv * w[t] + b[t];
    rowp[t] = fmaxf(y, 0.0f);
}

// ---------------------------------------------------------------------------
extern "C" void kernel_launch(void* const* d_in, const int* in_sizes, int n_in,
                              void* d_out, int out_size, void* d_ws, size_t ws_size,
                              hipStream_t stream) {
    const float*     node  = (const float*)d_in[0];
    const float*     node0 = (const float*)d_in[1];
    const long long* ei    = (const long long*)d_in[2];
    const float*     eattr = (const float*)d_in[3];
    // d_in[4] = batch_ptr (unused in node-mode LayerNorm)
    const float*     att_l = (const float*)d_in[5];
    const float*     att_r = (const float*)d_in[6];
    const float*     lnw   = (const float*)d_in[7];
    const float*     lnb   = (const float*)d_in[8];

    const int Nn = in_sizes[0] / D_FEAT;   // 50000
    const int E  = in_sizes[3];            // 800000
    float* out = (float*)d_out;

    float* al   = (float*)d_ws;
    float* ar   = al + Nn;
    float* coef = ar + Nn;

    // 1) out = 0.1 * node_0
    const int n4 = Nn * (D_FEAT / 4);
    fa_init_kernel<<<(n4 + 255) / 256, 256, 0, stream>>>((const float4*)node0, (float4*)out, n4);

    // 2) attention scalars via bf16x3 WMMA (one wave per 16-node tile)
    fa_att_kernel<<<(Nn + 15) / 16, 32, 0, stream>>>(node, att_l, att_r, al, ar, Nn);

    // 3) per-edge coefficients + L2 warm of gather rows
    fa_coef_kernel<<<(E + 255) / 256, 256, 0, stream>>>(ei, eattr, al, ar, node, coef, E);

    // 4) gather/scatter messages (64 threads per edge)
    long long threads = (long long)E * 64;
    fa_scatter_kernel<<<(int)((threads + 255) / 256), 256, 0, stream>>>(node, ei, coef, out, E);

    // 5) LayerNorm + ReLU in place
    fa_ln_relu_kernel<<<Nn, 256, 0, stream>>>(out, lnw, lnb);
}
